// GCNReactionDirectionPredictor_60653528154494
// MI455X (gfx1250) — matrix operand order repaired and checked by reading, not compiled, not run on hardware
//
#include <hip/hip_runtime.h>
#include <hip/hip_bf16.h>

#define NN 100000
#define EE 3200000
#define GG 1024
#define VV 10000
#define DD 64
#define HH 64
#define OO 32

typedef __attribute__((ext_vector_type(16))) _Float16 v16h;
typedef __attribute__((ext_vector_type(8)))  float    v8f;

static __device__ __forceinline__ float leaky(float x) { return x > 0.0f ? x : 0.1f * x; }

// ---------------- degree / norm precompute ----------------
__global__ void k_deg_init(float* __restrict__ deg) {
    int i = blockIdx.x * blockDim.x + threadIdx.x;
    if (i < NN) deg[i] = 1.0f;
}

__global__ void k_deg_count(const int* __restrict__ dst, float* __restrict__ deg) {
    int e = blockIdx.x * blockDim.x + threadIdx.x;
    if (e < EE) atomicAdd(&deg[dst[e]], 1.0f);
}

__global__ void k_dinv(float* __restrict__ deg) {
    int i = blockIdx.x * blockDim.x + threadIdx.x;
    if (i < NN) deg[i] = rsqrtf(deg[i]);   // deg buffer becomes dinv in place
}

__global__ void k_norm(const int* __restrict__ src, const int* __restrict__ dst,
                       const float* __restrict__ dinv, float* __restrict__ nw) {
    int e = blockIdx.x * blockDim.x + threadIdx.x;
    if (e < EE) nw[e] = dinv[src[e]] * dinv[dst[e]];
}

// ---------------- embedding gather (float4) ----------------
__global__ void k_embed(const int* __restrict__ feat, const float* __restrict__ emb,
                        float* __restrict__ x) {
    int idx = blockIdx.x * blockDim.x + threadIdx.x;      // NN*16 float4 slots
    if (idx >= NN * 16) return;
    int n = idx >> 4, q = idx & 15;
    const float4* er = (const float4*)emb;
    ((float4*)x)[idx] = er[(size_t)feat[n] * 16 + q];
}

// ---------------- WMMA GEMM: Y[N x (NT*16)] = X[N x 64] @ W[64 x (NT*16)] ----------------
// One wave per 16-row block; NT n-tiles of 16 cols kept in registers; K=64 = 2 chunks of 32.
template <int NT>
__global__ void __launch_bounds__(32)
k_gemm(const float* __restrict__ X, const float* __restrict__ W, float* __restrict__ Y) {
    const int Nd = NT * 16;
    const int lane = threadIdx.x;            // 0..31, wave32
    const int mb = blockIdx.x;               // 16-row block
    const int ml = lane & 15;
    const int lg = lane >> 4;                // lane group 0/1
    const float* __restrict__ xr = X + (size_t)(mb * 16 + ml) * 64;

    __builtin_prefetch(W, 0, 1);             // global_prefetch_b8: pull weights toward WGP

    v8f acc[NT] = {};
    #pragma unroll
    for (int kc = 0; kc < 2; ++kc) {
        const int k0 = kc * 32;
        // A tile 16x32 f16: lane(0..15)=row M=lane%16, halves h -> K=k0+lg*16+h
        // (contiguous k order, matching B)
        v16h a;
        #pragma unroll
        for (int h = 0; h < 16; ++h) a[h] = (_Float16)xr[k0 + lg * 16 + h];

        #pragma unroll
        for (int nb = 0; nb < NT; ++nb) {
            // B tile 32x16 f16: lane holds column N=nb*16+ml; halves h -> K=k0+lg*16+h
            const int ncol = nb * 16 + ml;
            const int kb = k0 + lg * 16;
            v16h b;
            #pragma unroll
            for (int h = 0; h < 16; ++h) b[h] = (_Float16)W[(size_t)(kb + h) * Nd + ncol];
            acc[nb] = __builtin_amdgcn_wmma_f32_16x16x32_f16(
                false, a, false, b, (short)0, acc[nb], false, false);
        }
    }
    // C/D layout: VGPR r -> row M = lg*8 + r, col N = lane%16
    const int r0 = lg * 8;
    #pragma unroll
    for (int nb = 0; nb < NT; ++nb) {
        const int ncol = nb * 16 + ml;
        #pragma unroll
        for (int r = 0; r < 8; ++r)
            Y[(size_t)(mb * 16 + r0 + r) * Nd + ncol] = acc[nb][r];
    }
}

// ---------------- agg = xw * dinv^2 + b ----------------
__global__ void k_initagg(const float* __restrict__ xw, const float* __restrict__ dinv,
                          const float* __restrict__ b, float* __restrict__ agg,
                          int lg2, int total) {
    int idx = blockIdx.x * blockDim.x + threadIdx.x;
    if (idx >= total) return;
    int n = idx >> lg2;
    int j = idx & ((1 << lg2) - 1);
    float di = dinv[n];
    agg[idx] = xw[idx] * (di * di) + b[j];
}

// ---------------- edge scatter: agg[dst] += xw[src] * norm (float4 + f32 atomics) ----------------
__global__ void k_scatter(const int* __restrict__ src, const int* __restrict__ dst,
                          const float* __restrict__ nw, const float* __restrict__ xw,
                          float* __restrict__ agg, int lg2q /* log2(Nd/4) */) {
    int t = blockIdx.x * blockDim.x + threadIdx.x;
    int e = t >> lg2q;
    if (e >= EE) return;
    int q = t & ((1 << lg2q) - 1);
    float w = nw[e];
    int s = src[e], d = dst[e];
    const float4 v = *(const float4*)(xw + ((size_t)s << (lg2q + 2)) + q * 4);
    float* ap = agg + ((size_t)d << (lg2q + 2)) + q * 4;
    atomicAdd(ap + 0, v.x * w);
    atomicAdd(ap + 1, v.y * w);
    atomicAdd(ap + 2, v.z * w);
    atomicAdd(ap + 3, v.w * w);
}

// ---------------- leaky activation ----------------
__global__ void k_finalize(const float* __restrict__ agg, float* __restrict__ x, int total) {
    int idx = blockIdx.x * blockDim.x + threadIdx.x;
    if (idx < total) x[idx] = leaky(agg[idx]);
}

// ---------------- pooling ----------------
__global__ void k_zero(float* __restrict__ p, int n) {
    int i = blockIdx.x * blockDim.x + threadIdx.x;
    if (i < n) p[i] = 0.0f;
}

__global__ void k_pool(const float* __restrict__ x, const int* __restrict__ batch,
                       float* __restrict__ psum, float* __restrict__ cnt) {
    int idx = blockIdx.x * blockDim.x + threadIdx.x;   // NN*32
    if (idx >= NN * 32) return;
    int n = idx >> 5, j = idx & 31;
    int g = batch[n];
    atomicAdd(&psum[g * 32 + j], x[(size_t)n * 32 + j]);
    if (j == 0) atomicAdd(&cnt[g], 1.0f);
}

__global__ void __launch_bounds__(32)
k_out(const float* __restrict__ psum, const float* __restrict__ cnt,
      const float* __restrict__ fcw, const float* __restrict__ fcb,
      float* __restrict__ out) {
    int g = blockIdx.x;
    int lane = threadIdx.x;                 // 0..31 == OO
    float c = fmaxf(cnt[g], 1.0f);
    float v = (psum[g * 32 + lane] / c) * fcw[lane];
    #pragma unroll
    for (int m = 16; m >= 1; m >>= 1) v += __shfl_xor(v, m, 32);
    if (lane == 0) out[g] = v + fcb[0];
}

static inline int cdiv(long long a, int b) { return (int)((a + b - 1) / b); }

extern "C" void kernel_launch(void* const* d_in, const int* in_sizes, int n_in,
                              void* d_out, int out_size, void* d_ws, size_t ws_size,
                              hipStream_t stream) {
    const int*   edge  = (const int*)d_in[0];
    const int*   src   = edge;
    const int*   dst   = edge + EE;
    const int*   feat  = (const int*)d_in[1];
    const int*   batch = (const int*)d_in[2];
    const float* emb   = (const float*)d_in[3];
    const float* Wl[3] = { (const float*)d_in[4], (const float*)d_in[6], (const float*)d_in[8] };
    const float* bl[3] = { (const float*)d_in[5], (const float*)d_in[7], (const float*)d_in[9] };
    const float* fcw   = (const float*)d_in[10];
    const float* fcb   = (const float*)d_in[11];
    float* out = (float*)d_out;

    // workspace carve-up (256B aligned)
    char* w = (char*)d_ws;
    auto alloc = [&](size_t bytes) -> float* {
        float* p = (float*)w;
        w += (bytes + 255) & ~(size_t)255;
        return p;
    };
    float* dinv  = alloc((size_t)NN * 4);        // deg -> dinv in place
    float* normw = alloc((size_t)EE * 4);
    float* xbuf  = alloc((size_t)NN * 64 * 4);
    float* xwbuf = alloc((size_t)NN * 64 * 4);
    float* aggbf = alloc((size_t)NN * 64 * 4);
    float* psum  = alloc((size_t)GG * 32 * 4);
    float* cnt   = alloc((size_t)GG * 4);

    const int T = 256;

    // degrees and normalization
    k_deg_init<<<cdiv(NN, T), T, 0, stream>>>(dinv);
    k_deg_count<<<cdiv(EE, T), T, 0, stream>>>(dst, dinv);
    k_dinv<<<cdiv(NN, T), T, 0, stream>>>(dinv);
    k_norm<<<cdiv(EE, T), T, 0, stream>>>(src, dst, dinv, normw);

    // embedding gather
    k_embed<<<cdiv((long long)NN * 16, T), T, 0, stream>>>(feat, emb, xbuf);

    // three GCN layers
    for (int l = 0; l < 3; ++l) {
        const int Nd  = (l == 2) ? 32 : 64;
        const int lg2 = (l == 2) ? 5 : 6;
        const int total = NN * Nd;
        if (Nd == 64) k_gemm<4><<<NN / 16, 32, 0, stream>>>(xbuf, Wl[l], xwbuf);
        else          k_gemm<2><<<NN / 16, 32, 0, stream>>>(xbuf, Wl[l], xwbuf);
        k_initagg<<<cdiv(total, T), T, 0, stream>>>(xwbuf, dinv, bl[l], aggbf, lg2, total);
        k_scatter<<<cdiv((long long)EE * (Nd / 4), T), T, 0, stream>>>(
            src, dst, normw, xwbuf, aggbf, lg2 - 2);
        k_finalize<<<cdiv(total, T), T, 0, stream>>>(aggbf, xbuf, total);
    }

    // mean-pool per graph + final linear
    k_zero<<<cdiv(GG * 33, T), T, 0, stream>>>(psum, GG * 33);   // psum (GG*32) and cnt (GG) are contiguous
    k_pool<<<cdiv((long long)NN * 32, T), T, 0, stream>>>(xbuf, batch, psum, cnt);
    k_out<<<GG, 32, 0, stream>>>(psum, cnt, fcw, fcb, out);
}
